// CR8_reg_cond_mul_6_13975823582043
// MI455X (gfx1250) — matrix operand order repaired and checked by reading, not compiled, not run on hardware
//
#include <hip/hip_runtime.h>
#include <stdint.h>

// ---------------------------------------------------------------------------
// Fused classifier+regressor for MI455X (gfx1250, wave32)
//   - bf16 WMMA 16x16x32 for all 5 GEMM stages (incl. the gathered MLP turned
//     into a dense [128x256]x[256x256] GEMM + per-token column select)
//   - persistent workgroups, double-buffered GLOBAL_LOAD_ASYNC_TO_LDS_B128
//     prefetch of the x_in tile (ASYNCcnt-gated), overlapping HBM with WMMA
// ---------------------------------------------------------------------------

typedef __attribute__((ext_vector_type(16))) __bf16 v16bf;
typedef __attribute__((ext_vector_type(8)))  __bf16 v8bf;
typedef __attribute__((ext_vector_type(8)))  float  v8f;

#define NTOK   (128 * 2048)
#define NTILES (NTOK / 128)
#define NWG    512           // persistent workgroups
#define LDA    136           // bf16 row stride, 128-col LDS tiles (272B = 17*16B)
#define LDT    280           // bf16 row stride, 256-col LDS tiles (560B = 35*16B)
#define LDS_STG 132          // fp32 row stride for staging (528B = 33*16B)

// bf16 weight layout inside d_ws (element offsets)
#define OFF_B1 0             // cl1_w  [128][128] (O-major == B layout)
#define OFF_B2 16384         // cl2_w
#define OFF_BR 32768         // reg1_w
#define OFF_B3 49152         // cl3_w rows 0..127
#define OFF_W2 65536         // w2 transposed: [n=s*32+o][i] (256x256)

__device__ __forceinline__ float lrelu(float v) { return v >= 0.0f ? v : 0.01f * v; }

// A fragment (16x32 bf16), contiguous K order (matches B):
//   lanes 0-15 : M=lane,    K = kk+{0..15}
//   lanes 16-31: M=lane-16, K = kk+{16..31}
__device__ __forceinline__ v16bf load_a(const __bf16* M, int stride, int row,
                                        int kk, int hf) {
  const __bf16* p = M + row * stride + kk + hf * 16;
  v8bf lo = *(const v8bf*)p;
  v8bf hi = *(const v8bf*)(p + 8);
  return __builtin_shufflevector(lo, hi, 0, 1, 2, 3, 4, 5, 6, 7,
                                 8, 9, 10, 11, 12, 13, 14, 15);
}

// Wave computes a 16(row) x 128(col) strip: C += A(16xK) * B(Kx128).
// B row-major [n][K] bf16 in global: lane n reads 16 contiguous K (32B load).
template <int KDIM>
__device__ __forceinline__ void mm16x128(const __bf16* A, int as, int m0,
                                         int lm, int hf,
                                         const __bf16* B, int nbase,
                                         v8f acc[8]) {
#pragma unroll
  for (int kk = 0; kk < KDIM; kk += 32) {
    v16bf a = load_a(A, as, m0 + lm, kk, hf);
#pragma unroll
    for (int nt = 0; nt < 8; ++nt) {
      const __bf16* bp = B + (size_t)(nbase + nt * 16 + lm) * KDIM + kk + hf * 16;
      v16bf b = *(const v16bf*)bp;
      acc[nt] = __builtin_amdgcn_wmma_f32_16x16x32_bf16(
          false, a, false, b, (short)0, acc[nt], false, false);
    }
  }
}

// C layout: lane ln, VGPR r -> M = m0 + r + hf*8, N = nt*16 + lm
__device__ __forceinline__ void store_act(const v8f acc[8], const float* bias,
                                          int nbias, __bf16* C, int cs, int coff,
                                          int m0, int lm, int hf) {
#pragma unroll
  for (int nt = 0; nt < 8; ++nt) {
    float bn = bias[nbias + nt * 16 + lm];
#pragma unroll
    for (int r = 0; r < 8; ++r) {
      float v = lrelu(acc[nt][r] + bn);
      C[(m0 + r + hf * 8) * cs + coff + nt * 16 + lm] = (__bf16)v;
    }
  }
}

// Async-prefetch one wave's 16 tokens x 128 channels (fp32) into LDS staging,
// channel-major [c][t] so each lane moves 16 contiguous bytes.
// 16 x GLOBAL_LOAD_ASYNC_TO_LDS_B128 per wave, tracked by ASYNCcnt.
__device__ __forceinline__ void issue_tile_async(const float* __restrict__ x_in,
                                                 int tile, float* stage,
                                                 int lane, int m0) {
  const int n0 = tile * 128;
  const float* xb = x_in + (size_t)(n0 >> 11) * (128 * 2048) + (n0 & 2047);
  const int tq = 4 * (lane & 3);   // token quad within wave's 16 tokens
  const int cb = lane >> 2;        // channel sub-index 0..7
#pragma unroll
  for (int i = 0; i < 16; ++i) {
    int c = i * 8 + cb;
    const float* ga = xb + (size_t)c * 2048 + (m0 + tq);
    unsigned lds = (unsigned)(uintptr_t)(stage + c * LDS_STG + m0 + tq);
    asm volatile("global_load_async_to_lds_b128 %0, %1, off"
                 :: "v"(lds), "v"(ga) : "memory");
  }
}

// ---------------------------------------------------------------------------
// Prep: fp32 weights -> bf16 B-operand layouts in d_ws
// ---------------------------------------------------------------------------
__global__ void prep_weights(const float* __restrict__ cl1_w,
                             const float* __restrict__ cl2_w,
                             const float* __restrict__ cl3_w,
                             const float* __restrict__ reg1_w,
                             const float* __restrict__ w2,
                             __bf16* __restrict__ ws) {
  int i = blockIdx.x * 256 + threadIdx.x;
  if (i < 16384) {
    ws[OFF_B1 + i] = (__bf16)cl1_w[i];
  } else if (i < 32768) {
    int j = i - 16384; ws[OFF_B2 + j] = (__bf16)cl2_w[j];
  } else if (i < 49152) {
    int j = i - 32768; ws[OFF_BR + j] = (__bf16)reg1_w[j];
  } else if (i < 65536) {
    int j = i - 49152; ws[OFF_B3 + j] = (__bf16)cl3_w[j];  // rows 0..127
  } else if (i < 131072) {
    int j = i - 65536;           // j = n*256 + i_in
    int n = j >> 8, ii = j & 255;
    int s = n >> 5, o = n & 31;  // n = s*32 + o
    ws[OFF_W2 + j] = (__bf16)w2[(s * 256 + ii) * 32 + o];
  }
}

// ---------------------------------------------------------------------------
// Main fused kernel: persistent WGs; one 128-token tile per iteration
// ---------------------------------------------------------------------------
__global__ __launch_bounds__(256) void fused_main(
    const float* __restrict__ x_in, const float* __restrict__ cl1_b,
    const float* __restrict__ cl2_b, const float* __restrict__ cl3_w,
    const float* __restrict__ cl3_b, const float* __restrict__ reg1_b,
    const float* __restrict__ b2, const float* __restrict__ w3,
    const float* __restrict__ b3, const __bf16* __restrict__ ws,
    float* __restrict__ out) {
  __shared__ float  sStage[2][128 * LDS_STG];  // fp32 x_in staging (double buf)
  __shared__ __bf16 sAX[128 * LDA];            // bf16 x tile; reused as X2
  __shared__ __bf16 sTOK[128 * LDT];           // [r | h1] concat, 256 cols
  __shared__ __bf16 sHALL[128 * LDT];          // H_all (bias+lrelu applied)
  __shared__ int    sIND[128];

  const int tid  = threadIdx.x;
  const int wave = tid >> 5;
  const int lane = tid & 31;
  const int hf   = lane >> 4;
  const int lm   = lane & 15;
  const int m0   = wave * 16;  // this wave's 16 token-rows

  // Prefetch first tile into buffer 0 (16 async ops per wave outstanding).
  int tile0 = blockIdx.x;
  if (tile0 < NTILES) issue_tile_async(x_in, tile0, sStage[0], lane, m0);

  int it = 0;
  for (int tile = tile0; tile < NTILES; tile += NWG, ++it) {
    const int cur = it & 1;
    const int n0  = tile * 128;

    // Kick off next tile's async prefetch into the other buffer, then wait
    // for the current tile (per-wave in-order async: 16 ops still in flight).
    const int ntile = tile + NWG;
    if (ntile < NTILES) {
      issue_tile_async(x_in, ntile, sStage[1 - cur], lane, m0);
      asm volatile("s_wait_asynccnt 0x10" ::: "memory");
    } else {
      asm volatile("s_wait_asynccnt 0x0" ::: "memory");
    }

    // Convert own rows: staging [c][t] fp32 -> sAX [t][c] bf16
    {
      const float* st = sStage[cur];
      int t = m0 + lm;
#pragma unroll
      for (int i = 0; i < 64; ++i) {
        int c = 2 * i + hf;
        sAX[t * LDA + c] = (__bf16)st[c * LDS_STG + t];
      }
    }
    // No barrier needed: every GEMM reads only this wave's own rows.

    v8f acc[8];

    // GEMM1: h1 = lrelu(x * cl1^T + b1)  -> sTOK cols 128..255
#pragma unroll
    for (int t = 0; t < 8; ++t) acc[t] = (v8f)0.0f;
    mm16x128<128>(sAX, LDA, m0, lm, hf, ws + OFF_B1, 0, acc);
    store_act(acc, cl1_b, 0, sTOK, LDT, 128, m0, lm, hf);

    // GEMM_R: r = lrelu(x * reg1^T + br) -> sTOK cols 0..127
#pragma unroll
    for (int t = 0; t < 8; ++t) acc[t] = (v8f)0.0f;
    mm16x128<128>(sAX, LDA, m0, lm, hf, ws + OFF_BR, 0, acc);
    store_act(acc, reg1_b, 0, sTOK, LDT, 0, m0, lm, hf);

    // GEMM2: x2 = lrelu(h1 * cl2^T + b2c) -> sAX (reuse)
#pragma unroll
    for (int t = 0; t < 8; ++t) acc[t] = (v8f)0.0f;
    mm16x128<128>(sTOK + 128, LDT, m0, lm, hf, ws + OFF_B2, 0, acc);
    store_act(acc, cl2_b, 0, sAX, LDA, 0, m0, lm, hf);

    // GEMM3: logits = x2 * cl3^T + b3c ; per-token argmax over 128 classes
#pragma unroll
    for (int t = 0; t < 8; ++t) acc[t] = (v8f)0.0f;
    mm16x128<128>(sAX, LDA, m0, lm, hf, ws + OFF_B3, 0, acc);
    {
      float bc[8];
#pragma unroll
      for (int nt = 0; nt < 8; ++nt) bc[nt] = cl3_b[nt * 16 + lm];
#pragma unroll
      for (int r = 0; r < 8; ++r) {
        float bv = -3.4e38f; int bi = 0;
#pragma unroll
        for (int nt = 0; nt < 8; ++nt) {
          float v = acc[nt][r] + bc[nt];
          int ci = nt * 16 + lm;
          if (v > bv || (v == bv && ci < bi)) { bv = v; bi = ci; }
        }
        for (int m = 1; m < 16; m <<= 1) {  // reduce within 16-lane half
          float ov = __shfl_xor(bv, m, 32);
          int   oi = __shfl_xor(bi, m, 32);
          if (ov > bv || (ov == bv && oi < bi)) { bv = ov; bi = oi; }
        }
        if (lm == 0) sIND[m0 + r + hf * 8] = bi;
      }
    }

    // GEMM4: H_all = lrelu(tok[128x256] * W2cat^T + b2cat) -> sHALL
#pragma unroll
    for (int ph = 0; ph < 2; ++ph) {
#pragma unroll
      for (int t = 0; t < 8; ++t) acc[t] = (v8f)0.0f;
      mm16x128<256>(sTOK, LDT, m0, lm, hf, ws + OFF_W2, ph * 128, acc);
      store_act(acc, b2, ph * 128, sHALL, LDT, ph * 128, m0, lm, hf);
    }

    __syncthreads();  // final phase reads cross-wave LDS

    if (tid < 128) {
      // x_real: select superclass slice of H_all + tiny GEMV with w3[ind]
      int t   = tid;
      int ind = sIND[t];
      int s   = ind >> 4;  // CLASS_FACTOR = 16
      float reg = b3[ind];
      const float* w3r = w3 + ind * 32;
#pragma unroll
      for (int o = 0; o < 32; ++o)
        reg += (float)sHALL[t * LDT + s * 32 + o] * w3r[o];
      out[n0 + t] = ((float)ind + reg) * (1.0f / 128.0f);
    } else {
      // mask = lrelu(x2 . cl3_w[row 128] + cl3_b[128])
      int t = tid - 128;
      const float* wm = cl3_w + 128 * 128;
      float a = cl3_b[128];
#pragma unroll
      for (int c = 0; c < 128; ++c)
        a += (float)sAX[t * LDA + c] * wm[c];
      out[NTOK + n0 + t] = lrelu(a);
    }

    __syncthreads();  // protect final-phase reads from next tile's writes
  }
}

// ---------------------------------------------------------------------------
extern "C" void kernel_launch(void* const* d_in, const int* in_sizes, int n_in,
                              void* d_out, int out_size, void* d_ws,
                              size_t ws_size, hipStream_t stream) {
  const float* x_in   = (const float*)d_in[0];
  const float* cl1_w  = (const float*)d_in[1];
  const float* cl1_b  = (const float*)d_in[2];
  const float* cl2_w  = (const float*)d_in[3];
  const float* cl2_b  = (const float*)d_in[4];
  const float* cl3_w  = (const float*)d_in[5];
  const float* cl3_b  = (const float*)d_in[6];
  const float* reg1_w = (const float*)d_in[7];
  const float* reg1_b = (const float*)d_in[8];
  const float* w2     = (const float*)d_in[9];
  const float* b2     = (const float*)d_in[10];
  const float* w3     = (const float*)d_in[11];
  const float* b3     = (const float*)d_in[12];
  __bf16* ws = (__bf16*)d_ws;
  float* out = (float*)d_out;

  prep_weights<<<512, 256, 0, stream>>>(cl1_w, cl2_w, cl3_w, reg1_w, w2, ws);
  fused_main<<<NWG, 256, 0, stream>>>(x_in, cl1_b, cl2_b, cl3_w, cl3_b,
                                      reg1_b, b2, w3, b3, ws, out);
}
